// Model_85401129714201
// MI455X (gfx1250) — matrix operand order repaired and checked by reading, not compiled, not run on hardware
//
#include <hip/hip_runtime.h>
#include <hip/hip_bf16.h>
#include <math.h>

#define BATCH  32
#define KNN    20
#define NSAMP  64
#define RAD2   0.04f     // RADIUS^2
#define LEAKYA 0.2f

typedef _Float16 half16 __attribute__((ext_vector_type(16)));
typedef float    float8 __attribute__((ext_vector_type(8)));
typedef unsigned int uintv4 __attribute__((ext_vector_type(4)));
typedef int      intv8  __attribute__((ext_vector_type(8)));
typedef int      intv4  __attribute__((ext_vector_type(4)));

#if defined(__has_builtin)
#if __has_builtin(__builtin_amdgcn_tensor_load_to_lds) && __has_builtin(__builtin_amdgcn_s_wait_tensorcnt)
#define HAVE_TDM 1
#endif
#endif

// ---------------------------------------------------------------------------
// A-operand loaders for the WMMA GEMM (C[M,N] = A[M,K] * W[N,K]^T)
// ---------------------------------------------------------------------------
struct DenseA {
  const float* A; int lda;
  __device__ __forceinline__ float ld(int r, int c) const {
    return A[(size_t)r * lda + c];
  }
};

// EdgeConv A: row r = (b,n,j) edge, A(r,c) = h[b, nbr[b,n,j], c] - h[b,n,c]
struct EdgeA {
  const float* h; const int* nbr; int N; int C;
  __device__ __forceinline__ float ld(int r, int c) const {
    int j  = r % KNN;
    int bn = r / KNN;
    int b  = bn / N;
    int src = nbr[(size_t)bn * KNN + j];
    return h[((size_t)b * N + src) * C + c] - h[(size_t)bn * C + c];
  }
};

// SA grouping A: row r = (b,s,j); c<3 -> xyz offset, else gathered feature
struct GroupA {
  const float* xyz; const float* feat; const float* nxyz; const int* ball;
  int N; int S; int C;
  __device__ __forceinline__ float ld(int r, int c) const {
    int j  = r & (NSAMP - 1);
    int bs = r >> 6;
    int b  = bs / S;
    int src = ball[(size_t)bs * NSAMP + j];
    if (c < 3)
      return xyz[((size_t)b * N + src) * 3 + c] - nxyz[(size_t)bs * 3 + c];
    return feat[((size_t)b * N + src) * C + (c - 3)];
  }
};

// ---------------------------------------------------------------------------
// TDM: DMA one 64x32 fp32 weight tile (row stride K elements) into LDS with
// 1-DWORD padding every 32 DWORDs (-> LDS row stride 33 floats).
// D# built per CDNA5 ISA ch.8 (group0: count/lds/global/type, group1: dims).
// Uses the 6-arg clang-23 builtin form.
// ---------------------------------------------------------------------------
#ifdef HAVE_TDM
__device__ __forceinline__ void tdm_load_w_tile(const float* gsrc, unsigned lds_byte,
                                                int Krow, int rowsTotal) {
  unsigned long long ga = (unsigned long long)(size_t)gsrc;
  uintv4 g0;
  g0[0] = 1u;                                             // count=1, user D#
  g0[1] = lds_byte;                                       // lds_addr (bytes)
  g0[2] = (unsigned)(ga & 0xFFFFFFFFull);                 // global_addr[31:0]
  g0[3] = (unsigned)((ga >> 32) & 0x01FFFFFFull)          // global_addr[56:32]
          | (2u << 30);                                   // type=2 ("image")
  intv8 g1;
  g1[0] = (int)((2u << 16)          // data_size=4B, workgroup_mask=0
          | (1u << 20)              // pad_enable
          | (4u << 22));            // pad_interval: 2^(4+1)=32 DWORDs, pad_amount=0 (1 DWORD)
  g1[1] = (int)(((unsigned)Krow & 0xFFFFu) << 16);        // tensor_dim0[15:0] @63:48
  g1[2] = (int)((((unsigned)Krow >> 16) & 0xFFFFu)        // tensor_dim0[31:16]
          | (((unsigned)rowsTotal & 0xFFFFu) << 16));     // tensor_dim1[15:0]
  g1[3] = (int)((((unsigned)rowsTotal >> 16) & 0xFFFFu)   // tensor_dim1[31:16]
          | (32u << 16));                                 // tile_dim0 = 32
  g1[4] = 64;                                             // tile_dim1 = 64, tile_dim2 = 0
  g1[5] = Krow;                                           // tensor_dim0_stride[31:0]
  g1[6] = 0;                                              // stride hi / dim1_stride lo
  g1[7] = 0;
  intv4 z4 = {};
  intv8 z8 = {};
  __builtin_amdgcn_tensor_load_to_lds(g0, g1, z4, z4, z8, 0);
}
#endif

// ---------------------------------------------------------------------------
// WMMA GEMM: block = 128 threads (4 waves), tile 64(M) x 64(N), K step 32.
// Each wave owns a 32x32 quadrant: 2 A-frags x 2 B-frags -> 4 WMMAs/K-step.
// A staged fp32->f16 in LDS; W staged as raw fp32 (TDM fast path for full
// tiles), converted to f16 at fragment build.
// ---------------------------------------------------------------------------
template <class LA>
__global__ void wmma_gemm_nt(LA la, const float* __restrict__ W,
                             float* __restrict__ Cout,
                             int M, int N, int K, const float* __restrict__ bias) {
  __shared__ _Float16 As[64][36];
  __shared__ float    Wf[64][33];

  const int tid  = threadIdx.x;
  const int lane = tid & 31;
  const int wave = tid >> 5;
  const int wm   = (wave >> 1) << 5;   // 0 or 32
  const int wn   = (wave & 1) << 5;    // 0 or 32
  const int m0   = blockIdx.y << 6;
  const int n0   = blockIdx.x << 6;

  float8 acc[2][2] = {};

  for (int kt = 0; kt < K; kt += 32) {
    // --- stage A tile 64x32 (f32 -> f16) ---
    for (int i = tid; i < 64 * 32; i += 128) {
      int r = i >> 5, c = i & 31;
      int gm = m0 + r, gk = kt + c;
      float av = (gm < M && gk < K) ? la.ld(gm, gk) : 0.0f;
      As[r][c] = (_Float16)av;
    }
    // --- stage W tile 64x32 fp32 ---
    bool full = (n0 + 64 <= N) && (kt + 32 <= K);
#ifdef HAVE_TDM
    if (full) {
      if (wave == 0) {
        tdm_load_w_tile(W + (size_t)n0 * K + kt,
                        (unsigned)(size_t)&Wf[0][0], K, N);
        __builtin_amdgcn_s_wait_tensorcnt(0);
      }
    } else
#endif
    {
      for (int i = tid; i < 64 * 32; i += 128) {
        int r = i >> 5, c = i & 31;
        int gn = n0 + r, gk = kt + c;
        Wf[r][c] = (gn < N && gk < K) ? W[(size_t)gn * K + gk] : 0.0f;
      }
      if (kt + 32 < K) {
        int gn = n0 + lane;
        if (gn < N) __builtin_prefetch(W + (size_t)gn * K + (kt + 32), 0, 0);
      }
    }
    __syncthreads();

    // --- build fragments: contiguous k = 16*half + s in both A and B ---
    const int half_ = lane >> 4;
    half16 af[2], bf[2];
#pragma unroll
    for (int mi = 0; mi < 2; ++mi) {
      int mrow = wm + (mi << 4) + (lane & 15);
#pragma unroll
      for (int r = 0; r < 8; ++r) {
        int ka = 16 * half_ + 2 * r;
        af[mi][2 * r]     = As[mrow][ka];
        af[mi][2 * r + 1] = As[mrow][ka + 1];
      }
    }
#pragma unroll
    for (int ni = 0; ni < 2; ++ni) {
      int ncol = wn + (ni << 4) + (lane & 15);
#pragma unroll
      for (int r = 0; r < 8; ++r) {
        int kb = 16 * half_ + 2 * r;
        bf[ni][2 * r]     = (_Float16)Wf[ncol][kb];
        bf[ni][2 * r + 1] = (_Float16)Wf[ncol][kb + 1];
      }
    }
#pragma unroll
    for (int mi = 0; mi < 2; ++mi)
#pragma unroll
      for (int ni = 0; ni < 2; ++ni)
        acc[mi][ni] = __builtin_amdgcn_wmma_f32_16x16x32_f16(
            false, af[mi], false, bf[ni], (short)0, acc[mi][ni], false, false);
    __syncthreads();
  }

  // --- epilogue: C layout: n = lane&15, m = r + 8*(lane>>4) ---
#pragma unroll
  for (int mi = 0; mi < 2; ++mi)
#pragma unroll
    for (int ni = 0; ni < 2; ++ni)
#pragma unroll
      for (int r = 0; r < 8; ++r) {
        int m = m0 + wm + (mi << 4) + r + ((lane >> 4) << 3);
        int n = n0 + wn + (ni << 4) + (lane & 15);
        if (m < M && n < N) {
          float v = acc[mi][ni][r];
          if (bias) v += bias[n];
          Cout[(size_t)m * N + n] = v;
        }
      }
}

// ---------------------------------------------------------------------------
// KNN: one thread per (b,n), insertion-sorted top-20 smallest squared dist.
// ---------------------------------------------------------------------------
__global__ void knn_kernel(const float* __restrict__ xyz, int N, int* __restrict__ nbr) {
  int t = blockIdx.x * blockDim.x + threadIdx.x;
  if (t >= BATCH * N) return;
  int b = t / N, n = t % N;
  const float* p = xyz + (size_t)b * N * 3;
  float qx = p[n * 3 + 0], qy = p[n * 3 + 1], qz = p[n * 3 + 2];
  float kd[KNN]; int ki[KNN];
#pragma unroll
  for (int i = 0; i < KNN; ++i) { kd[i] = 3.4e38f; ki[i] = 0; }
  for (int m = 0; m < N; ++m) {
    float dx = p[m * 3 + 0] - qx;
    float dy = p[m * 3 + 1] - qy;
    float dz = p[m * 3 + 2] - qz;
    float d = dx * dx + dy * dy + dz * dz;
    if (d < kd[KNN - 1]) {
      int j = KNN - 1;
      while (j > 0 && kd[j - 1] > d) { kd[j] = kd[j - 1]; ki[j] = ki[j - 1]; --j; }
      kd[j] = d; ki[j] = m;
    }
  }
  for (int i = 0; i < KNN; ++i) nbr[(size_t)t * KNN + i] = ki[i];
}

// ---------------------------------------------------------------------------
// BN stats: one block per channel; e = E[r,o] (+ P[r/knn,o]); writes mean/invstd.
// ---------------------------------------------------------------------------
__global__ void bn_stats(const float* __restrict__ E, const float* __restrict__ P,
                         int rows, int knn, int O,
                         float* __restrict__ mean, float* __restrict__ invstd) {
  int o = blockIdx.x;
  float s = 0.0f, s2 = 0.0f;
  const bool hasP = (P != nullptr);
  for (int r = threadIdx.x; r < rows; r += blockDim.x) {
    float v = E[(size_t)r * O + o];
    if (hasP) v += P[(size_t)(r / knn) * O + o];
    s += v; s2 += v * v;
  }
  __shared__ float rs[256], rq[256];
  rs[threadIdx.x] = s; rq[threadIdx.x] = s2;
  __syncthreads();
  for (int st = 128; st > 0; st >>= 1) {
    if ((int)threadIdx.x < st) {
      rs[threadIdx.x] += rs[threadIdx.x + st];
      rq[threadIdx.x] += rq[threadIdx.x + st];
    }
    __syncthreads();
  }
  if (threadIdx.x == 0) {
    float mu = rs[0] / (float)rows;
    float var = rq[0] / (float)rows - mu * mu;
    mean[o] = mu;
    invstd[o] = rsqrtf(var + 1e-5f);
  }
}

// EdgeConv epilogue: BN -> max over k -> leaky relu
__global__ void edge_apply_max(const float* __restrict__ E, const float* __restrict__ P,
                               const float* __restrict__ mean, const float* __restrict__ invstd,
                               const float* __restrict__ bns, const float* __restrict__ bnb,
                               int BN_, int O, float* __restrict__ hout) {
  int t = blockIdx.x * blockDim.x + threadIdx.x;
  if (t >= BN_ * O) return;
  int bn = t / O, o = t % O;
  float p = P[(size_t)bn * O + o];
  float mu = mean[o], is = invstd[o], sc = bns[o], bi = bnb[o];
  float best = -3.4e38f;
  for (int j = 0; j < KNN; ++j) {
    float v = E[((size_t)bn * KNN + j) * O + o] + p;
    v = (v - mu) * is * sc + bi;
    best = fmaxf(best, v);
  }
  hout[t] = best > 0.0f ? best : LEAKYA * best;
}

// BN + ReLU in place over [rows, O]
__global__ void bn_relu(float* __restrict__ G,
                        const float* __restrict__ mean, const float* __restrict__ invstd,
                        const float* __restrict__ s, const float* __restrict__ b,
                        size_t total, int O) {
  size_t t = (size_t)blockIdx.x * blockDim.x + threadIdx.x;
  if (t >= total) return;
  int o = (int)(t % O);
  float v = (G[t] - mean[o]) * invstd[o] * s[o] + b[o];
  G[t] = fmaxf(v, 0.0f);
}

// max over nsample: new_feat[bs,o] = max_j G[(bs*64+j), o]
__global__ void maxpool_ns(const float* __restrict__ G, int BS, int O,
                           float* __restrict__ nf) {
  int t = blockIdx.x * blockDim.x + threadIdx.x;
  if (t >= BS * O) return;
  int bs = t / O, o = t % O;
  float best = -3.4e38f;
  for (int j = 0; j < NSAMP; ++j)
    best = fmaxf(best, G[((size_t)bs * NSAMP + j) * O + o]);
  nf[t] = best;
}

// ---------------------------------------------------------------------------
// Furthest point sampling: one block per batch element.
// ---------------------------------------------------------------------------
__global__ void fps_kernel(const float* __restrict__ xyz, int N, int S,
                           int* __restrict__ out) {
  __shared__ float dist[1024];
  __shared__ float rmax[256];
  __shared__ int   rid[256];
  __shared__ int   last;
  int b = blockIdx.x;
  int tid = threadIdx.x;
  const float* p = xyz + (size_t)b * N * 3;
  for (int n = tid; n < N; n += 256) dist[n] = 1e10f;
  if (tid == 0) { last = 0; out[(size_t)b * S] = 0; }
  __syncthreads();
  for (int t = 1; t < S; ++t) {
    float lx = p[last * 3 + 0], ly = p[last * 3 + 1], lz = p[last * 3 + 2];
    float bm = -1.0f; int bi = 0;
    for (int n = tid; n < N; n += 256) {
      float dx = p[n * 3 + 0] - lx;
      float dy = p[n * 3 + 1] - ly;
      float dz = p[n * 3 + 2] - lz;
      float d = dx * dx + dy * dy + dz * dz;
      float nd = fminf(dist[n], d);
      dist[n] = nd;
      if (nd > bm) { bm = nd; bi = n; }
    }
    rmax[tid] = bm; rid[tid] = bi;
    __syncthreads();
    for (int st = 128; st > 0; st >>= 1) {
      if (tid < st && rmax[tid + st] > rmax[tid]) {
        rmax[tid] = rmax[tid + st]; rid[tid] = rid[tid + st];
      }
      __syncthreads();
    }
    if (tid == 0) { last = rid[0]; out[(size_t)b * S + t] = rid[0]; }
    __syncthreads();
  }
}

__global__ void gather_newxyz(const float* __restrict__ xyz, const int* __restrict__ idx,
                              int N, int S, float* __restrict__ nxyz) {
  int t = blockIdx.x * blockDim.x + threadIdx.x;
  if (t >= BATCH * S) return;
  int b = t / S;
  int id = idx[t];
  const float* src = xyz + ((size_t)b * N + id) * 3;
  nxyz[(size_t)t * 3 + 0] = src[0];
  nxyz[(size_t)t * 3 + 1] = src[1];
  nxyz[(size_t)t * 3 + 2] = src[2];
}

__global__ void ballquery(const float* __restrict__ xyz, const float* __restrict__ nxyz,
                          int N, int S, int* __restrict__ ball) {
  int t = blockIdx.x * blockDim.x + threadIdx.x;
  if (t >= BATCH * S) return;
  int b = t / S;
  float qx = nxyz[(size_t)t * 3 + 0];
  float qy = nxyz[(size_t)t * 3 + 1];
  float qz = nxyz[(size_t)t * 3 + 2];
  const float* p = xyz + (size_t)b * N * 3;
  int* o = ball + (size_t)t * NSAMP;
  int cnt = 0;
  for (int n = 0; n < N && cnt < NSAMP; ++n) {
    float dx = p[n * 3 + 0] - qx;
    float dy = p[n * 3 + 1] - qy;
    float dz = p[n * 3 + 2] - qz;
    if (dx * dx + dy * dy + dz * dz < RAD2) o[cnt++] = n;
  }
  int pad = (cnt > 0) ? o[0] : 0;
  for (; cnt < NSAMP; ++cnt) o[cnt] = pad;
}

// ---------------------------------------------------------------------------
// SE block
// ---------------------------------------------------------------------------
__global__ void se_reduce(const float* __restrict__ nf, int S, int O,
                          float* __restrict__ sm) {
  int t = blockIdx.x * blockDim.x + threadIdx.x;
  if (t >= BATCH * O) return;
  int b = t / O, o = t % O;
  float s = 0.0f;
  for (int j = 0; j < S; ++j) s += nf[((size_t)b * S + j) * O + o];
  sm[t] = s / (float)S;
}

__global__ void se_fc(const float* __restrict__ sm, const float* __restrict__ se1,
                      const float* __restrict__ se2, int O, int mid,
                      float* __restrict__ scale) {
  __shared__ float sv[512];
  __shared__ float mv[64];
  int b = blockIdx.x;
  int tid = threadIdx.x;
  sv[tid] = sm[(size_t)b * O + tid];
  __syncthreads();
  if (tid < mid) {
    float a = 0.0f;
    for (int c = 0; c < O; ++c) a += se1[(size_t)tid * O + c] * sv[c];
    mv[tid] = fmaxf(a, 0.0f);
  }
  __syncthreads();
  float a = 0.0f;
  for (int m = 0; m < mid; ++m) a += se2[(size_t)tid * mid + m] * mv[m];
  scale[(size_t)b * O + tid] = 1.0f + 1.0f / (1.0f + expf(-a));
}

__global__ void se_scale(float* __restrict__ nf, const float* __restrict__ scale,
                         int S, int O) {
  int t = blockIdx.x * blockDim.x + threadIdx.x;
  if (t >= BATCH * S * O) return;
  int o = t % O;
  int b = t / (S * O);
  nf[t] *= scale[(size_t)b * O + o];
}

// ---------------------------------------------------------------------------
// Final head
// ---------------------------------------------------------------------------
__global__ void pool_maxavg(const float* __restrict__ h, int S, int C,
                            float* __restrict__ pooled) {
  int t = blockIdx.x * blockDim.x + threadIdx.x;
  if (t >= BATCH * C) return;
  int b = t / C, c = t % C;
  float mx = -3.4e38f, s = 0.0f;
  for (int j = 0; j < S; ++j) {
    float v = h[((size_t)b * S + j) * C + c];
    mx = fmaxf(mx, v); s += v;
  }
  pooled[(size_t)b * (2 * C) + c] = mx;
  pooled[(size_t)b * (2 * C) + C + c] = s / (float)S;
}

__global__ void bn_batch(float* __restrict__ h, const float* __restrict__ s,
                         const float* __restrict__ b_, int rows, int C) {
  int t = blockIdx.x * blockDim.x + threadIdx.x;
  if (t >= C) return;
  float mu = 0.0f, q = 0.0f;
  for (int r = 0; r < rows; ++r) {
    float v = h[(size_t)r * C + t];
    mu += v; q += v * v;
  }
  mu /= (float)rows;
  float var = q / (float)rows - mu * mu;
  float is = rsqrtf(var + 1e-5f);
  for (int r = 0; r < rows; ++r)
    h[(size_t)r * C + t] = (h[(size_t)r * C + t] - mu) * is * s[t] + b_[t];
}

// ---------------------------------------------------------------------------
// Host orchestration
// ---------------------------------------------------------------------------
extern "C" void kernel_launch(void* const* d_in, const int* in_sizes, int n_in,
                              void* d_out, int out_size, void* d_ws, size_t ws_size,
                              hipStream_t stream) {
  (void)in_sizes; (void)n_in; (void)out_size; (void)ws_size;
  const float* xyz_in = (const float*)d_in[0];
  const float* rgb    = (const float*)d_in[1];
  auto prm = [&](int i) { return (const float*)d_in[i]; };

  static const int Ns[5]      = {1024, 512, 256, 128, 64};
  static const int CinArr[4]  = {3, 64, 128, 256};
  static const int CoutArr[4] = {64, 128, 256, 512};

  char* wp = (char*)d_ws;
  auto carve = [&](size_t bytes) -> char* {
    char* p = wp;
    wp += (bytes + 255) & ~(size_t)255;
    return p;
  };
  float* st_mean = (float*)carve(512 * sizeof(float));
  float* st_inv  = (float*)carve(512 * sizeof(float));
  int*   nbr     = (int*)carve((size_t)BATCH * 1024 * KNN * sizeof(int));
  int*   fpsidx  = (int*)carve((size_t)BATCH * 512 * sizeof(int));
  int*   ball    = (int*)carve((size_t)BATCH * 512 * NSAMP * sizeof(int));
  float* xyzb0   = (float*)carve((size_t)BATCH * 512 * 3 * sizeof(float));
  float* xyzb1   = (float*)carve((size_t)BATCH * 512 * 3 * sizeof(float));
  float* hb0     = (float*)carve((size_t)2097152 * sizeof(float));
  float* hb1     = (float*)carve((size_t)2097152 * sizeof(float));
  float* hbe     = (float*)carve((size_t)2097152 * sizeof(float));   // edgeconv out
  float* Pbuf    = (float*)carve((size_t)2097152 * sizeof(float));
  float* smean   = (float*)carve((size_t)BATCH * 512 * sizeof(float));
  float* sscale  = (float*)carve((size_t)BATCH * 512 * sizeof(float));
  float* pooled  = (float*)carve((size_t)BATCH * 1024 * sizeof(float));
  float* hemb    = (float*)carve((size_t)BATCH * 512 * sizeof(float));
  float* GA      = (float*)carve((size_t)67108864 * sizeof(float));  // 268 MB arena
  float* GB      = (float*)carve((size_t)67108864 * sizeof(float));  // 268 MB arena

  const float* xyz_cur = xyz_in;
  const float* h_cur   = rgb;

  for (int i = 0; i < 4; ++i) {
    const int N = Ns[i], S = Ns[i + 1];
    const int Ci = CinArr[i], Co = CoutArr[i];
    const int base = 2 + i * 15;
    const float* theta = prm(base + 0);
    const float* phi   = prm(base + 1);
    const float* bns   = prm(base + 2);
    const float* bnb   = prm(base + 3);
    const float* m0w = prm(base + 4),  *m0s = prm(base + 5),  *m0b = prm(base + 6);
    const float* m1w = prm(base + 7),  *m1s = prm(base + 8),  *m1b = prm(base + 9);
    const float* m2w = prm(base + 10), *m2s = prm(base + 11), *m2b = prm(base + 12);
    const float* se1 = prm(base + 13), *se2 = prm(base + 14);

    // --- KNN graph ---
    knn_kernel<<<(BATCH * N + 127) / 128, 128, 0, stream>>>(xyz_cur, N, nbr);

    // --- EdgeConv: E1 = (h_j - h_i) @ theta^T (gather fused), P = h @ phi^T ---
    const int Me = BATCH * N * KNN;
    float* E1 = GA;  // alias: dead before SA-MLP phase reuses GA
    {
      dim3 g((Co + 63) / 64, (Me + 63) / 64);
      EdgeA la{h_cur, nbr, N, Ci};
      wmma_gemm_nt<EdgeA><<<g, 128, 0, stream>>>(la, theta, E1, Me, Co, Ci, nullptr);
    }
    {
      dim3 g((Co + 63) / 64, (BATCH * N + 63) / 64);
      DenseA la{h_cur, Ci};
      wmma_gemm_nt<DenseA><<<g, 128, 0, stream>>>(la, phi, Pbuf, BATCH * N, Co, Ci, nullptr);
    }
    bn_stats<<<Co, 256, 0, stream>>>(E1, Pbuf, Me, KNN, Co, st_mean, st_inv);
    edge_apply_max<<<(BATCH * N * Co + 255) / 256, 256, 0, stream>>>(
        E1, Pbuf, st_mean, st_inv, bns, bnb, BATCH * N, Co, hbe);

    // --- SA module: FPS -> ball query -> 3x (GEMM + BN + ReLU) -> maxpool -> SE ---
    float* nxyz = (i & 1) ? xyzb1 : xyzb0;
    fps_kernel<<<BATCH, 256, 0, stream>>>(xyz_cur, N, S, fpsidx);
    gather_newxyz<<<(BATCH * S + 127) / 128, 128, 0, stream>>>(xyz_cur, fpsidx, N, S, nxyz);
    ballquery<<<(BATCH * S + 127) / 128, 128, 0, stream>>>(xyz_cur, nxyz, N, S, ball);

    const int Mg = BATCH * S * NSAMP;
    {  // MLP layer 0: K = Co + 3, A gathered on the fly
      dim3 g((Co + 63) / 64, (Mg + 63) / 64);
      GroupA la{xyz_cur, hbe, nxyz, ball, N, S, Co};
      wmma_gemm_nt<GroupA><<<g, 128, 0, stream>>>(la, m0w, GA, Mg, Co, Co + 3, nullptr);
    }
    bn_stats<<<Co, 256, 0, stream>>>(GA, nullptr, Mg, 1, Co, st_mean, st_inv);
    bn_relu<<<(int)(((size_t)Mg * Co + 255) / 256), 256, 0, stream>>>(
        GA, st_mean, st_inv, m0s, m0b, (size_t)Mg * Co, Co);
    {  // MLP layer 1
      dim3 g((Co + 63) / 64, (Mg + 63) / 64);
      DenseA la{GA, Co};
      wmma_gemm_nt<DenseA><<<g, 128, 0, stream>>>(la, m1w, GB, Mg, Co, Co, nullptr);
    }
    bn_stats<<<Co, 256, 0, stream>>>(GB, nullptr, Mg, 1, Co, st_mean, st_inv);
    bn_relu<<<(int)(((size_t)Mg * Co + 255) / 256), 256, 0, stream>>>(
        GB, st_mean, st_inv, m1s, m1b, (size_t)Mg * Co, Co);
    {  // MLP layer 2
      dim3 g((Co + 63) / 64, (Mg + 63) / 64);
      DenseA la{GB, Co};
      wmma_gemm_nt<DenseA><<<g, 128, 0, stream>>>(la, m2w, GA, Mg, Co, Co, nullptr);
    }
    bn_stats<<<Co, 256, 0, stream>>>(GA, nullptr, Mg, 1, Co, st_mean, st_inv);
    bn_relu<<<(int)(((size_t)Mg * Co + 255) / 256), 256, 0, stream>>>(
        GA, st_mean, st_inv, m2s, m2b, (size_t)Mg * Co, Co);

    float* nfeat = (i & 1) ? hb1 : hb0;
    maxpool_ns<<<(BATCH * S * Co + 255) / 256, 256, 0, stream>>>(GA, BATCH * S, Co, nfeat);

    const int mid = (Co / 16 > 4) ? Co / 16 : 4;
    se_reduce<<<(BATCH * Co + 255) / 256, 256, 0, stream>>>(nfeat, S, Co, smean);
    se_fc<<<BATCH, Co, 0, stream>>>(smean, se1, se2, Co, mid, sscale);
    se_scale<<<(BATCH * S * Co + 255) / 256, 256, 0, stream>>>(nfeat, sscale, S, Co);

    h_cur = nfeat;
    xyz_cur = nxyz;
  }

  // --- head: max/avg pool -> emb GEMM -> batch BN -> proj GEMM + bias ---
  pool_maxavg<<<(BATCH * 512 + 255) / 256, 256, 0, stream>>>(h_cur, 64, 512, pooled);
  {
    dim3 g((512 + 63) / 64, (BATCH + 63) / 64);
    DenseA la{pooled, 1024};
    wmma_gemm_nt<DenseA><<<g, 128, 0, stream>>>(la, prm(62), hemb, BATCH, 512, 1024, nullptr);
  }
  bn_batch<<<2, 256, 0, stream>>>(hemb, prm(63), prm(64), BATCH, 512);
  {
    dim3 g((751 + 63) / 64, (BATCH + 63) / 64);
    DenseA la{hemb, 512};
    wmma_gemm_nt<DenseA><<<g, 128, 0, stream>>>(la, prm(65), (float*)d_out,
                                                BATCH, 751, 512, prm(66));
  }
}
